// _MultiHeadAttention_8186207666453
// MI455X (gfx1250) — matrix operand order repaired and checked by reading, not compiled, not run on hardware
//
#include <hip/hip_runtime.h>
#include <hip/hip_bf16.h>

#define BS   2
#define SEQ  2048
#define DM   1024
#define NH   16
#define DK   64
#define DVV  64

typedef _Float16 v16h __attribute__((ext_vector_type(16)));
typedef _Float16 v8h  __attribute__((ext_vector_type(8)));
typedef float    v8f  __attribute__((ext_vector_type(8)));
typedef float    v4f  __attribute__((ext_vector_type(4)));

static __device__ __forceinline__ v16h cat8(v8h lo, v8h hi) {
  return __builtin_shufflevector(lo, hi, 0,1,2,3,4,5,6,7,8,9,10,11,12,13,14,15);
}

static __device__ __forceinline__ v8f wmma16(v16h a, v16h b, v8f c) {
  // D = A(16x32 f16) * B(32x16 f16) + C(16x16 f32)
  return __builtin_amdgcn_wmma_f32_16x16x32_f16(false, a, false, b, (short)0, c, false, false);
}

// A fragment (16x32, f16). rowPtr = &A[lane's M row][0]; K chunk base k0.
static __device__ __forceinline__ v16h load_a_f16(const _Float16* rowPtr, int k0, int laneHi) {
  int kb = laneHi * 8;
  v8h lo = *(const v8h*)(rowPtr + k0 + kb);
  v8h hi = *(const v8h*)(rowPtr + k0 + 16 + kb);
  return cat8(lo, hi);
}

// B fragment (32x16, f16) from N-major (transposed) storage: rowPtr = &BT[lane's N][0].
static __device__ __forceinline__ v16h load_b_f16(const _Float16* rowPtr, int k0, int laneHi) {
  const _Float16* p = rowPtr + k0 + laneHi * 8;
  v8h lo = *(const v8h*)(p);
  v8h hi = *(const v8h*)(p + 16);
  return cat8(lo, hi);
}

static __device__ __forceinline__ v16h load_a_f32(const float* rowPtr, int k0, int laneHi) {
  int kb = laneHi * 8;
  v4f x0 = *(const v4f*)(rowPtr + k0 + kb);
  v4f x1 = *(const v4f*)(rowPtr + k0 + kb + 4);
  v4f x2 = *(const v4f*)(rowPtr + k0 + 16 + kb);
  v4f x3 = *(const v4f*)(rowPtr + k0 + 16 + kb + 4);
  v16h r;
#pragma unroll
  for (int i = 0; i < 4; ++i) {
    r[i]      = (_Float16)x0[i];
    r[4 + i]  = (_Float16)x1[i];
    r[8 + i]  = (_Float16)x2[i];
    r[12 + i] = (_Float16)x3[i];
  }
  return r;
}

static __device__ __forceinline__ v16h load_b_f32(const float* rowPtr, int k0, int laneHi) {
  const float* p = rowPtr + k0 + laneHi * 8;
  v16h r;
#pragma unroll
  for (int i = 0; i < 16; i += 4) {
    v4f x = *(const v4f*)(p + i + (i >= 8 ? 8 : 0));
    r[i] = (_Float16)x[0]; r[i+1] = (_Float16)x[1];
    r[i+2] = (_Float16)x[2]; r[i+3] = (_Float16)x[3];
  }
  return r;
}

// Async global->LDS stage of one contiguous 2KB tile (CDNA5 ASYNCcnt path).
// Each lane moves 16B per instruction; 4 instructions cover 2048B.
static __device__ __forceinline__ void stage_async_2k(unsigned ldsBase, const void* gsrc, int lane) {
  const char* g = (const char*)gsrc + lane * 16;
  unsigned    l = ldsBase + lane * 16;
#pragma unroll
  for (int i = 0; i < 4; ++i) {
    asm volatile("global_load_async_to_lds_b128 %0, %1, off"
                 :: "v"(l + i * 512), "v"(g + i * 512)
                 : "memory");
  }
}
static __device__ __forceinline__ void wait_async_le4() {
  asm volatile("s_wait_asynccnt 0x4" ::: "memory");
}
static __device__ __forceinline__ void wait_async_0() {
  asm volatile("s_wait_asynccnt 0x0" ::: "memory");
}

// Map projection element (r, o) of P = X @ W^T into attention-friendly layouts.
static __device__ __forceinline__ size_t proj_index(int matId, int r, int o) {
  int b  = r >> 11;
  int rr = r & 2047;
  int h  = rr >> 7;
  int t  = rr & 127;
  size_t head = (size_t)(b * NH + h);
  if (matId == 0) {       // QS[b,h,q,d]
    int q = t * 16 + (o >> 6);
    int d = o & 63;
    return (head * SEQ + q) * DK + d;
  }
  if (matId == 1) {       // KT[b,h,kcol,d]
    int kcol = (t & 1) * 1024 + o;
    int d = t >> 1;
    return (head * SEQ + kcol) * DK + d;
  }
  // VT[b,h,v,krow]
  int krow = t * 16 + (o >> 6);
  int v = o & 63;
  return (head * DVV + v) * SEQ + krow;
}

// ---------------- Kernel 1: projection GEMM (X[4096,1024] @ W^T) -> f16 scatter
__global__ __launch_bounds__(256) void proj_kernel(const float* __restrict__ X,
                                                   const float* __restrict__ W,
                                                   _Float16* __restrict__ dst,
                                                   int matId) {
  int tid = threadIdx.x;
  int lane = tid & 31, wave = tid >> 5;
  int laneLo = lane & 15, laneHi = lane >> 4;
  int task = blockIdx.x * 8 + wave;
  int mt = task >> 4, ng = task & 15;
  int r0 = mt * 16, n0 = ng * 64;

  const float* aRow  = X + (size_t)(r0 + laneLo) * DM;
  const float* bRow0 = W + (size_t)(n0 + laneLo) * DM;

  v8f c0 = {}, c1 = {}, c2 = {}, c3 = {};
  for (int kk = 0; kk < DM; kk += 32) {
    v16h a  = load_a_f32(aRow, kk, laneHi);
    if (kk + 64 < DM) __builtin_prefetch(aRow + kk + 64, 0, 1);
    v16h b0 = load_b_f32(bRow0,           kk, laneHi);
    v16h b1 = load_b_f32(bRow0 + 16 * DM, kk, laneHi);
    v16h b2 = load_b_f32(bRow0 + 32 * DM, kk, laneHi);
    v16h b3 = load_b_f32(bRow0 + 48 * DM, kk, laneHi);
    c0 = wmma16(a, b0, c0);
    c1 = wmma16(a, b1, c1);
    c2 = wmma16(a, b2, c2);
    c3 = wmma16(a, b3, c3);
  }
#pragma unroll
  for (int t = 0; t < 4; ++t) {
    v8f c = (t == 0) ? c0 : (t == 1) ? c1 : (t == 2) ? c2 : c3;
    int o = n0 + t * 16 + laneLo;
#pragma unroll
    for (int v = 0; v < 8; ++v) {
      int r = r0 + v + 8 * laneHi;
      dst[proj_index(matId, r, o)] = (_Float16)c[v];
    }
  }
}

// ---------------- Kernel 2: fused scores -> softmax -> attn store -> context
__global__ __launch_bounds__(256) void attn_kernel(const _Float16* __restrict__ QS,
                                                   const _Float16* __restrict__ KT,
                                                   const _Float16* __restrict__ VT,
                                                   _Float16* __restrict__ CT,
                                                   float* __restrict__ attnOut) {
  __shared__ float rowsum[16];
  __shared__ __align__(16) _Float16 stage[8][16][32];      // 8KB attn f16 re-layout
  __shared__ float ctxAcc[16][64];                         // 4KB
  __shared__ __align__(16) _Float16 ktile[8][2][16 * 64];  // 32KB double-buffered K tiles

  int tid = threadIdx.x;
  int lane = tid & 31, wave = tid >> 5;
  int laneLo = lane & 15, laneHi = lane >> 4;
  int bid = blockIdx.x;
  int qb = bid & 127;
  int h  = (bid >> 7) & 15;
  int b  = bid >> 11;
  int q0 = qb * 16;
  size_t head = (size_t)(b * NH + h);

  if (tid < 16) rowsum[tid] = 0.f;
  for (int i = tid; i < 16 * 64; i += 256) (&ctxAcc[0][0])[i] = 0.f;
  __syncthreads();

  const _Float16* aRow = QS + (head * SEQ + q0 + laneLo) * DK;
  v16h a0 = load_a_f16(aRow, 0,  laneHi);
  v16h a1 = load_a_f16(aRow, 32, laneHi);
  const float scale = 0.125f;

  const _Float16* ktHead = KT + head * SEQ * DK;           // tile kt starts at kt*16*64
  unsigned ldsBuf[2];
  ldsBuf[0] = (unsigned)(uintptr_t)&ktile[wave][0][0];
  ldsBuf[1] = (unsigned)(uintptr_t)&ktile[wave][1][0];

  // ---- sweep 1: row sums of exp(scores); KT staged via async-to-LDS, double buffered
  float rsum[8];
#pragma unroll
  for (int v = 0; v < 8; ++v) rsum[v] = 0.f;

  stage_async_2k(ldsBuf[0], ktHead + (size_t)(wave * 16) * 16 * DK, lane);
  for (int j = 0; j < 16; ++j) {
    if (j + 1 < 16) {
      stage_async_2k(ldsBuf[(j + 1) & 1],
                     ktHead + (size_t)(wave * 16 + j + 1) * 16 * DK, lane);
      wait_async_le4();     // tile j complete (loads retire in order)
    } else {
      wait_async_0();
    }
    const _Float16* bRow = (const _Float16*)(&ktile[wave][j & 1][0]) + laneLo * DK;
    v16h b0 = load_b_f16(bRow, 0,  laneHi);
    v16h b1 = load_b_f16(bRow, 32, laneHi);
    v8f c = {};
    c = wmma16(a0, b0, c);
    c = wmma16(a1, b1, c);
#pragma unroll
    for (int v = 0; v < 8; ++v) rsum[v] += __expf(c[v] * scale);
  }
#pragma unroll
  for (int v = 0; v < 8; ++v) {
    rsum[v] += __shfl_xor(rsum[v], 1, 16);
    rsum[v] += __shfl_xor(rsum[v], 2, 16);
    rsum[v] += __shfl_xor(rsum[v], 4, 16);
    rsum[v] += __shfl_xor(rsum[v], 8, 16);
  }
  if (laneLo == 0) {
#pragma unroll
    for (int v = 0; v < 8; ++v) atomicAdd(&rowsum[v + 8 * laneHi], rsum[v]);
  }
  __syncthreads();

  float rinv[8];
#pragma unroll
  for (int v = 0; v < 8; ++v) rinv[v] = 1.0f / rowsum[v + 8 * laneHi];

  // ---- sweep 2: recompute, normalize, stream attn (NT stores), accumulate attn @ V
  float* attnBase = attnOut + (head * SEQ + q0) * (size_t)SEQ;
  v8f ctx0 = {}, ctx1 = {}, ctx2 = {}, ctx3 = {};

  stage_async_2k(ldsBuf[0], ktHead + (size_t)(wave * 16) * 16 * DK, lane);
  for (int j = 0; j < 16; ++j) {
    int kt = wave * 16 + j;
    if (j + 1 < 16) {
      stage_async_2k(ldsBuf[(j + 1) & 1],
                     ktHead + (size_t)(kt + 1) * 16 * DK, lane);
      wait_async_le4();
    } else {
      wait_async_0();
    }
    const _Float16* bRow = (const _Float16*)(&ktile[wave][j & 1][0]) + laneLo * DK;
    v16h b0 = load_b_f16(bRow, 0,  laneHi);
    v16h b1 = load_b_f16(bRow, 32, laneHi);
    v8f c = {};
    c = wmma16(a0, b0, c);
    c = wmma16(a1, b1, c);
    int col = kt * 16 + laneLo;
#pragma unroll
    for (int v = 0; v < 8; ++v) {
      float e = __expf(c[v] * scale) * rinv[v];
      __builtin_nontemporal_store(e, &attnBase[(size_t)(v + 8 * laneHi) * SEQ + col]);
      stage[wave][v + 8 * laneHi][(j & 1) * 16 + laneLo] = (_Float16)e;
    }
    if (j & 1) {
      int kbase = wave * 256 + (j - 1) * 16;
      v16h aa = load_a_f16(&stage[wave][laneLo][0], 0, laneHi);
      const _Float16* vRow = VT + (head * DVV + laneLo) * SEQ;
      v16h bv0 = load_b_f16(vRow,            kbase, laneHi);
      v16h bv1 = load_b_f16(vRow + 16 * SEQ, kbase, laneHi);
      v16h bv2 = load_b_f16(vRow + 32 * SEQ, kbase, laneHi);
      v16h bv3 = load_b_f16(vRow + 48 * SEQ, kbase, laneHi);
      ctx0 = wmma16(aa, bv0, ctx0);
      ctx1 = wmma16(aa, bv1, ctx1);
      ctx2 = wmma16(aa, bv2, ctx2);
      ctx3 = wmma16(aa, bv3, ctx3);
    }
  }

  // cross-wave context reduction (ds_add_f32)
#pragma unroll
  for (int v = 0; v < 8; ++v) {
    int row = v + 8 * laneHi;
    atomicAdd(&ctxAcc[row][ 0 + laneLo], ctx0[v]);
    atomicAdd(&ctxAcc[row][16 + laneLo], ctx1[v]);
    atomicAdd(&ctxAcc[row][32 + laneLo], ctx2[v]);
    atomicAdd(&ctxAcc[row][48 + laneLo], ctx3[v]);
  }
  __syncthreads();

  for (int i = tid; i < 16 * 64; i += 256) {
    int row = i >> 6, col = i & 63;
    CT[(size_t)(b * SEQ + q0 + row) * DM + h * DVV + col] = (_Float16)ctxAcc[row][col];
  }
}

// ---------------- Kernel 3: output GEMM  out = CT[4096,1024] @ W_O^T, fp32 out
__global__ __launch_bounds__(256) void out_kernel(const _Float16* __restrict__ CT,
                                                  const float* __restrict__ WO,
                                                  float* __restrict__ out) {
  int tid = threadIdx.x;
  int lane = tid & 31, wave = tid >> 5;
  int laneLo = lane & 15, laneHi = lane >> 4;
  int task = blockIdx.x * 8 + wave;
  int mt = task >> 4, ng = task & 15;
  int r0 = mt * 16, n0 = ng * 64;

  const _Float16* aRow = CT + (size_t)(r0 + laneLo) * DM;
  const float*    bRow = WO + (size_t)(n0 + laneLo) * DM;

  v8f c0 = {}, c1 = {}, c2 = {}, c3 = {};
  for (int kk = 0; kk < DM; kk += 32) {
    v16h a  = load_a_f16(aRow, kk, laneHi);
    if (kk + 128 < DM) __builtin_prefetch(aRow + kk + 128, 0, 1);
    v16h b0 = load_b_f32(bRow,           kk, laneHi);
    v16h b1 = load_b_f32(bRow + 16 * DM, kk, laneHi);
    v16h b2 = load_b_f32(bRow + 32 * DM, kk, laneHi);
    v16h b3 = load_b_f32(bRow + 48 * DM, kk, laneHi);
    c0 = wmma16(a, b0, c0);
    c1 = wmma16(a, b1, c1);
    c2 = wmma16(a, b2, c2);
    c3 = wmma16(a, b3, c3);
  }
#pragma unroll
  for (int t = 0; t < 4; ++t) {
    v8f c = (t == 0) ? c0 : (t == 1) ? c1 : (t == 2) ? c2 : c3;
#pragma unroll
    for (int v = 0; v < 8; ++v) {
      __builtin_nontemporal_store(
          c[v], &out[(size_t)(r0 + v + 8 * laneHi) * DM + n0 + t * 16 + laneLo]);
    }
  }
}

extern "C" void kernel_launch(void* const* d_in, const int* in_sizes, int n_in,
                              void* d_out, int out_size, void* d_ws, size_t ws_size,
                              hipStream_t stream) {
  (void)in_sizes; (void)n_in; (void)out_size; (void)ws_size;
  const float* Q  = (const float*)d_in[0];
  const float* K  = (const float*)d_in[1];
  const float* V  = (const float*)d_in[2];
  const float* WQ = (const float*)d_in[3];
  const float* WK = (const float*)d_in[4];
  const float* WV = (const float*)d_in[5];
  const float* WO = (const float*)d_in[6];

  _Float16* QS = (_Float16*)d_ws;
  _Float16* KT = QS + (size_t)BS * NH * SEQ * DK;
  _Float16* VT = KT + (size_t)BS * NH * SEQ * DK;
  _Float16* CT = VT + (size_t)BS * NH * DVV * SEQ;

  float* out  = (float*)d_out;
  float* attn = out + (size_t)BS * SEQ * DM;

  dim3 blk(256);
  proj_kernel<<<512, blk, 0, stream>>>(Q, WQ, QS, 0);
  proj_kernel<<<512, blk, 0, stream>>>(K, WK, KT, 1);
  proj_kernel<<<512, blk, 0, stream>>>(V, WV, VT, 2);
  attn_kernel<<<BS * NH * (SEQ / 16), blk, 0, stream>>>(QS, KT, VT, CT, attn);
  out_kernel<<<512, blk, 0, stream>>>(CT, WO, out);
}
